// PointerNetworkDecoder_30631706755213
// MI455X (gfx1250) — compile-verified, matrix operand order checked by reading, not run on hardware
//
#include <hip/hip_runtime.h>
#include <hip/hip_bf16.h>

// ---------------------------------------------------------------------------
// Pointer-network decoder for MI455X (gfx1250, wave32, WMMA f16 16x16x32,
// Tensor Data Mover for LDS staging).
// B = N = T = 192, D = 512, H = 8, DH = 64.
// ---------------------------------------------------------------------------

typedef _Float16 h16  __attribute__((ext_vector_type(16)));
typedef _Float16 h8   __attribute__((ext_vector_type(8)));
typedef _Float16 h2   __attribute__((ext_vector_type(2)));
typedef float    f32x8 __attribute__((ext_vector_type(8)));
typedef unsigned int u32x4 __attribute__((ext_vector_type(4)));
typedef int      i32x4 __attribute__((ext_vector_type(4)));
typedef int      i32x8 __attribute__((ext_vector_type(8)));

#define Bdim 192
#define Ndim 192
#define Tdim 192
#define Ddim 512
#define Hh   8
#define DH   64
#define G4D  2048      // 4*D
#define XK   1024      // [dec|h] inner dim
#define NEGV (-1.0e9f)

#define NBLK 96
#define BLKT 256
#define GSZ  (NBLK * BLKT)      // 24576 threads
#define TW   (GSZ / 32)         // 768 waves

// ---- workspace layout (bytes) --------------------------------------------
constexpr size_t OFF_BAR   = 0;                                            // barrier
constexpr size_t OFF_WCAT  = 256;                                          // [2048][1024] f16
constexpr size_t OFF_WQ    = OFF_WCAT  + (size_t)G4D * XK * 2;             // [512][512] f16
constexpr size_t OFF_WK    = OFF_WQ    + (size_t)Ddim * Ddim * 2;          // [512][512] f16
constexpr size_t OFF_WV    = OFF_WK    + (size_t)Ddim * Ddim * 2;          // [512][512] f16
constexpr size_t OFF_WOUTS = OFF_WV    + (size_t)Ddim * Ddim * 2;          // 512 f32 + bsum
constexpr size_t OFF_ENCH  = OFF_WOUTS + 2304;                             // [B*N][512] f16
constexpr size_t OFF_KH    = OFF_ENCH  + (size_t)Bdim * Ndim * Ddim * 2;   // [B][H][N][DH] f16
constexpr size_t OFF_VH    = OFF_KH    + (size_t)Bdim * Hh * Ndim * DH * 2;
constexpr size_t OFF_X     = OFF_VH    + (size_t)Bdim * Hh * Ndim * DH * 2; // [B][1024] f16
constexpr size_t OFF_GATES = OFF_X     + (size_t)Bdim * XK * 2;            // [B][2048] f32
constexpr size_t OFF_C     = OFF_GATES + (size_t)Bdim * G4D * 4;           // [B][512] f32
constexpr size_t OFF_Q     = OFF_C     + (size_t)Bdim * Ddim * 4;          // [B][512] f32
constexpr size_t OFF_ATTN  = OFF_Q     + (size_t)Bdim * Ddim * 4;          // [B][512] f32
constexpr size_t OFF_WW    = OFF_ATTN  + (size_t)Bdim * Ddim * 4;          // [B][H][N] f32
constexpr size_t OFF_LOG   = OFF_WW    + (size_t)Bdim * Hh * Ndim * 4;     // [B] f32
constexpr size_t OFF_VIS   = OFF_LOG   + 1024;                             // [B][N] u32

// ---- WMMA fragment load: two contiguous b128 chunks per lane --------------
// CDNA5 16-bit A/B frag: lane<16 holds K = k0+[0..7] (VGPR0-3) and k0+16+[0..7]
// (VGPR4-7); lane>=16 the same shifted by +8.
__device__ __forceinline__ h16 load_frag(const _Float16* __restrict__ rowp,
                                         int k0, int lane) {
  const _Float16* p = rowp + k0 + ((lane & 16) >> 1);
  h8 c0 = *(const h8*)(p);
  h8 c1 = *(const h8*)(p + 16);
  return __builtin_shufflevector(c0, c1, 0, 1, 2, 3, 4, 5, 6, 7,
                                 8, 9, 10, 11, 12, 13, 14, 15);
}

__device__ __forceinline__ f32x8 wmma_f16(h16 a, h16 b, f32x8 c) {
  return __builtin_amdgcn_wmma_f32_16x16x32_f16(false, a, false, b, (short)0, c,
                                                false, false);
}

// D = A * W^T over one 16x16 tile; A f16 [M][lda], W f16 row-major [N][ldw]
__device__ __forceinline__ f32x8 gemm_tile(const _Float16* __restrict__ A, int lda,
                                           const _Float16* __restrict__ W, int ldw,
                                           int mtile, int ntile, int Kdim, int lane) {
  f32x8 acc = {};
  const _Float16* arow = A + (size_t)(mtile * 16 + (lane & 15)) * lda;
  const _Float16* brow = W + (size_t)(ntile * 16 + (lane & 15)) * ldw;
  for (int k0 = 0; k0 < Kdim; k0 += 32) {
    acc = wmma_f16(load_frag(arow, k0, lane), load_frag(brow, k0, lane), acc);
  }
  return acc;
}

// ---- Tensor Data Mover: load a contiguous [rows][row_halves] f16 tile ----
#define HAS_TDM __has_builtin(__builtin_amdgcn_tensor_load_to_lds)

__device__ __forceinline__ void load_tile_to_lds(_Float16* lds_tile,
                                                 const _Float16* gsrc,
                                                 unsigned rows,
                                                 unsigned row_halves) {
#if HAS_TDM
  if (threadIdx.x < 32) {   // one wave issues the DMA (EXEC ignored by TDM)
    unsigned long long ga = (unsigned long long)(size_t)gsrc;
    unsigned lds_off =
        (unsigned)(unsigned long long)(__attribute__((address_space(3))) _Float16*)lds_tile;
    u32x4 g0;
    g0[0] = 1u;                                        // count=1 (user D#)
    g0[1] = lds_off;                                   // lds_addr
    g0[2] = (unsigned)(ga & 0xFFFFFFFFull);            // global_addr[31:0]
    g0[3] = ((unsigned)(ga >> 32) & 0x01FFFFFFu) | 0x80000000u;  // addr[56:32]|type=2
    i32x8 g1;
    g1[0] = 1 << 16;                                   // data_size = 2 bytes
    g1[1] = (int)((row_halves & 0xFFFFu) << 16);       // tensor_dim0[15:0]
    g1[2] = (int)((row_halves >> 16) | ((rows & 0xFFFFu) << 16)); // dim0 hi | dim1 lo
    g1[3] = (int)((rows >> 16) | (row_halves << 16));  // dim1 hi | tile_dim0
    g1[4] = (int)rows;                                 // tile_dim1 (tile_dim2 = 0)
    g1[5] = (int)row_halves;                           // tensor_dim0_stride[31:0]
    g1[6] = 0;
    g1[7] = 0;
    i32x4 z4 = {};
#if defined(__clang_major__) && (__clang_major__ >= 23)
    i32x8 z8 = {};
    __builtin_amdgcn_tensor_load_to_lds(g0, g1, z4, z4, z8, 0);
#else
    __builtin_amdgcn_tensor_load_to_lds(g0, g1, z4, z4, 0);
#endif
    __builtin_amdgcn_s_wait_tensorcnt(0);
  }
#else
  for (unsigned i = threadIdx.x * 8; i < rows * row_halves; i += blockDim.x * 8) {
    *(h8*)(lds_tile + i) = *(const h8*)(gsrc + i);
  }
#endif
  __syncthreads();
}

// ---- grid-wide barrier (sense reversing, persistent kernel) ---------------
__device__ __forceinline__ void grid_barrier(unsigned* cnt, unsigned* gen) {
  __syncthreads();
  if (threadIdx.x == 0) {
    __threadfence();
    unsigned old = __hip_atomic_load(gen, __ATOMIC_ACQUIRE, __HIP_MEMORY_SCOPE_AGENT);
    if (__hip_atomic_fetch_add(cnt, 1u, __ATOMIC_ACQ_REL, __HIP_MEMORY_SCOPE_AGENT) ==
        (unsigned)(NBLK - 1)) {
      __hip_atomic_store(cnt, 0u, __ATOMIC_RELAXED, __HIP_MEMORY_SCOPE_AGENT);
      __hip_atomic_fetch_add(gen, 1u, __ATOMIC_ACQ_REL, __HIP_MEMORY_SCOPE_AGENT);
    } else {
      while (__hip_atomic_load(gen, __ATOMIC_ACQUIRE, __HIP_MEMORY_SCOPE_AGENT) == old)
        __builtin_amdgcn_s_sleep(2);
    }
  }
  __syncthreads();
}

__device__ __forceinline__ float sigf(float x) { return 1.0f / (1.0f + __expf(-x)); }

__device__ __forceinline__ float gumbel_hash(unsigned t, unsigned r, unsigned n) {
  unsigned x = t * 0x9E3779B9u ^ (r * 0x85EBCA6Bu + 0x165667B1u) ^ (n * 0xC2B2AE35u);
  x ^= x >> 16; x *= 0x7FEB352Du; x ^= x >> 15; x *= 0x846CA68Bu; x ^= x >> 16;
  float u = (float)(x >> 8) * (1.0f / 16777216.0f) + 1.0e-10f;
  return -__logf(-__logf(u));
}

// ===========================================================================
// Prep kernels: f16 conversions + logits weight folding.
// ===========================================================================
__global__ void prep_weights(char* __restrict__ ws,
                             const float* __restrict__ W_ih,
                             const float* __restrict__ W_hh,
                             const float* __restrict__ in_proj_w) {
  _Float16* Wcat = (_Float16*)(ws + OFF_WCAT);
  _Float16* Wq   = (_Float16*)(ws + OFF_WQ);
  _Float16* Wk   = (_Float16*)(ws + OFF_WK);
  _Float16* Wv   = (_Float16*)(ws + OFF_WV);
  const int stride = gridDim.x * blockDim.x;
  for (int i = blockIdx.x * blockDim.x + threadIdx.x; i < G4D * XK; i += stride) {
    int n = i >> 10, k = i & (XK - 1);
    float v = (k < Ddim) ? W_ih[(size_t)n * Ddim + k]
                         : W_hh[(size_t)n * Ddim + (k - Ddim)];
    Wcat[i] = (_Float16)v;
  }
  for (int i = blockIdx.x * blockDim.x + threadIdx.x; i < Ddim * Ddim; i += stride) {
    Wq[i] = (_Float16)in_proj_w[i];
    Wk[i] = (_Float16)in_proj_w[Ddim * Ddim + i];
    Wv[i] = (_Float16)in_proj_w[2 * Ddim * Ddim + i];
  }
}

__global__ void prep_enc(char* __restrict__ ws, const float* __restrict__ enc) {
  _Float16* encH = (_Float16*)(ws + OFF_ENCH);
  const int total = Bdim * Ndim * Ddim;
  for (int i = blockIdx.x * blockDim.x + threadIdx.x; i < total;
       i += gridDim.x * blockDim.x)
    encH[i] = (_Float16)enc[i];
}

__global__ void prep_woutsum(char* __restrict__ ws,
                             const float* __restrict__ Wout,
                             const float* __restrict__ bout) {
  float* wsum = (float*)(ws + OFF_WOUTS);
  int k = threadIdx.x;
  if (k < Ddim) {
    float s = 0.0f;
    for (int d = 0; d < Ddim; ++d) s += Wout[(size_t)d * Ddim + k];
    wsum[k] = s;
  }
  if (k == 0) {
    float s = 0.0f;
    for (int d = 0; d < Ddim; ++d) s += bout[d];
    wsum[Ddim] = s;
  }
}

// ===========================================================================
// K/V projection GEMM (WMMA, all-f16): [36864,512] x [512,512] twice.
// Output layout [b][h][n][dh] f16 for attention locality.
// ===========================================================================
__global__ void kv_proj(char* __restrict__ ws, const float* __restrict__ in_proj_b) {
  const _Float16* encH = (const _Float16*)(ws + OFF_ENCH);
  const int lane = threadIdx.x & 31;
  const int gw   = (blockIdx.x * blockDim.x + threadIdx.x) >> 5;
  const int nwaves = (gridDim.x * blockDim.x) >> 5;
  const int MT = (Bdim * Ndim) / 16;  // 2304
  const int NT = Ddim / 16;           // 32
  const int tiles_per = MT * NT;      // 73728 per projection

  for (int tile = gw; tile < 2 * tiles_per; tile += nwaves) {
    int sel   = tile / tiles_per;     // 0 = K, 1 = V
    int t     = tile % tiles_per;
    int mtile = t / NT;
    int ntile = t % NT;
    const _Float16* W = (const _Float16*)(ws + (sel ? OFF_WV : OFF_WK));
    f32x8 acc = gemm_tile(encH, Ddim, W, Ddim, mtile, ntile, Ddim, lane);

    _Float16* dst = (_Float16*)(ws + (sel ? OFF_VH : OFF_KH));
    const float* bias = in_proj_b + (sel ? 1024 : 512);
    int n  = ntile * 16 + (lane & 15);
    int hh = n >> 6, dh = n & 63;
    int bb = mtile / (Ndim / 16);                 // 16-row tile never crosses b
    int nk0 = (mtile % (Ndim / 16)) * 16 + ((lane & 16) ? 8 : 0);
    float bval = bias[n];
    _Float16* base = dst + (((size_t)bb * Hh + hh) * Ndim + nk0) * DH + dh;
#pragma unroll
    for (int v = 0; v < 8; ++v) base[(size_t)v * DH] = (_Float16)(acc[v] + bval);
  }
}

// ===========================================================================
// Persistent sequential decoder (192 steps, global barriers, TDM staging).
// ===========================================================================
__global__ void decode_persistent(char* __restrict__ ws,
                                  const float* __restrict__ enc,
                                  const float* __restrict__ b_ih,
                                  const float* __restrict__ b_hh,
                                  const float* __restrict__ in_proj_b,
                                  float* __restrict__ out) {
  unsigned* cnt = (unsigned*)(ws + OFF_BAR);
  unsigned* gen = cnt + 1;
  _Float16* Wcat = (_Float16*)(ws + OFF_WCAT);
  _Float16* Wq   = (_Float16*)(ws + OFF_WQ);
  float*    wsum = (float*)(ws + OFF_WOUTS);
  _Float16* Kh   = (_Float16*)(ws + OFF_KH);
  _Float16* Vh   = (_Float16*)(ws + OFF_VH);
  _Float16* X    = (_Float16*)(ws + OFF_X);
  float*    gates = (float*)(ws + OFF_GATES);
  float*    Cst   = (float*)(ws + OFF_C);
  float*    Qst   = (float*)(ws + OFF_Q);
  float*    attn  = (float*)(ws + OFF_ATTN);
  float*    wwgt  = (float*)(ws + OFF_WW);
  float*    logit = (float*)(ws + OFF_LOG);
  unsigned* vis   = (unsigned*)(ws + OFF_VIS);
  float*    seq_out = out;                         // [B][T]
  float*    aw_out  = out + (size_t)Bdim * Tdim;   // [B][T][N]

  __shared__ __align__(128) _Float16 Atile[16 * XK];   // 32 KB per block

  const int tid  = blockIdx.x * blockDim.x + threadIdx.x;
  const int lane = threadIdx.x & 31;
  const int wv   = threadIdx.x >> 5;               // wave in block (0..7)
  const int gw   = tid >> 5;

  // ---- phase 0: init (dec = mean(enc), h = 0, c = 0, visited = 0) ----
  for (int i = tid; i < Bdim * Ddim; i += GSZ) {
    int b = i >> 9, d = i & (Ddim - 1);
    float s = 0.0f;
    for (int n = 0; n < Ndim; ++n) s += enc[((size_t)b * Ndim + n) * Ddim + d];
    X[(size_t)b * XK + d]        = (_Float16)(s * (1.0f / (float)Ndim));
    X[(size_t)b * XK + Ddim + d] = (_Float16)0.0f;
    Cst[i] = 0.0f;
  }
  for (int i = tid; i < Bdim * Ndim; i += GSZ) vis[i] = 0u;
  grid_barrier(cnt, gen);

  const int mtile_blk = blockIdx.x % (Bdim / 16);  // 0..11
  const int nchunk    = blockIdx.x / (Bdim / 16);  // 0..7

  for (int t = 0; t < Tdim; ++t) {
    // ---- phase 1: gates = X @ Wcat^T + bias   [192 x 2048] ----
    // block owns one 16-row A-tile of X: TDM it into LDS once, reuse 16x.
    load_tile_to_lds(Atile, X + (size_t)mtile_blk * 16 * XK, 16, XK);
    {
      const _Float16* arow = Atile + (size_t)(lane & 15) * XK;
#pragma unroll
      for (int j = 0; j < 2; ++j) {
        int ntile = nchunk * 16 + wv * 2 + j;
        const _Float16* brow = Wcat + (size_t)(ntile * 16 + (lane & 15)) * XK;
        f32x8 acc = {};
        for (int k0 = 0; k0 < XK; k0 += 32)
          acc = wmma_f16(load_frag(arow, k0, lane), load_frag(brow, k0, lane), acc);
        int n = ntile * 16 + (lane & 15);
        float bias = b_ih[n] + b_hh[n];
        int m0 = mtile_blk * 16 + ((lane & 16) ? 8 : 0);
#pragma unroll
        for (int v = 0; v < 8; ++v)
          gates[(size_t)(m0 + v) * G4D + n] = acc[v] + bias;
      }
    }
    grid_barrier(cnt, gen);

    // ---- phase 2: LSTM elementwise -> c, h (h stored f16 into X) ----
    for (int i = tid; i < Bdim * Ddim; i += GSZ) {
      int b = i >> 9, d = i & (Ddim - 1);
      const float* g = gates + (size_t)b * G4D;
      float ig = sigf(g[d]);
      float fg = sigf(g[Ddim + d]);
      float gg = tanhf(g[2 * Ddim + d]);
      float og = sigf(g[3 * Ddim + d]);
      float cn = fg * Cst[i] + ig * gg;
      float hn = og * tanhf(cn);
      Cst[i] = cn;
      X[(size_t)b * XK + Ddim + d] = (_Float16)hn;
    }
    grid_barrier(cnt, gen);

    // ---- phase 3: q = (h @ Wq^T + bq) * 1/sqrt(DH)  [192 x 512] ----
    for (int tile = gw; tile < (Bdim / 16) * (Ddim / 16); tile += TW) {
      int mtile = tile >> 5;          // Ddim/16 == 32
      int ntile = tile & 31;
      f32x8 acc = gemm_tile(X + Ddim, XK, Wq, Ddim, mtile, ntile, Ddim, lane);
      int n = ntile * 16 + (lane & 15);
      float bq = in_proj_b[n];
      int m0 = mtile * 16 + ((lane & 16) ? 8 : 0);
#pragma unroll
      for (int v = 0; v < 8; ++v)
        Qst[(size_t)(m0 + v) * Ddim + n] = (acc[v] + bq) * 0.125f;  // 1/sqrt(64)
    }
    grid_barrier(cnt, gen);

    // ---- phase 4: attention per (b,h): scores, softmax, attn = w@V ----
    for (int bh = gw; bh < Bdim * Hh; bh += TW) {
      int b = bh >> 3, h = bh & 7;
      const float*    qv = Qst + (size_t)b * Ddim + h * DH;
      const _Float16* Kb = Kh + ((size_t)bh) * Ndim * DH;
      const _Float16* Vb = Vh + ((size_t)bh) * Ndim * DH;
      float sc[6];
#pragma unroll
      for (int j = 0; j < 6; ++j) {
        int n = lane + 32 * j;
        const _Float16* kr = Kb + (size_t)n * DH;
        __builtin_prefetch(kr + DH, 0, 1);
        float s = 0.0f;
        for (int d2 = 0; d2 < DH / 2; ++d2) {
          h2 kk = *(const h2*)(kr + 2 * d2);
          s += qv[2 * d2] * (float)kk.x + qv[2 * d2 + 1] * (float)kk.y;
        }
        sc[j] = s;
      }
      float mx = sc[0];
#pragma unroll
      for (int j = 1; j < 6; ++j) mx = fmaxf(mx, sc[j]);
      for (int off = 16; off; off >>= 1) mx = fmaxf(mx, __shfl_xor(mx, off, 32));
      float sum = 0.0f, wl[6];
#pragma unroll
      for (int j = 0; j < 6; ++j) { wl[j] = __expf(sc[j] - mx); sum += wl[j]; }
      for (int off = 16; off; off >>= 1) sum += __shfl_xor(sum, off, 32);
      float inv = 1.0f / sum;
#pragma unroll
      for (int j = 0; j < 6; ++j) {
        wl[j] *= inv;
        wwgt[(size_t)bh * Ndim + lane + 32 * j] = wl[j];
      }
      float a0 = 0.0f, a1 = 0.0f;
      for (int n = 0; n < Ndim; ++n) {
        float wn = __shfl(wl[n >> 5], n & 31, 32);
        h2 vv = *(const h2*)(Vb + (size_t)n * DH + 2 * lane);
        a0 += wn * (float)vv.x;
        a1 += wn * (float)vv.y;
      }
      attn[(size_t)b * Ddim + h * DH + 2 * lane]     = a0;
      attn[(size_t)b * Ddim + h * DH + 2 * lane + 1] = a1;
    }
    grid_barrier(cnt, gen);

    // ---- phase 5: logits = attn . colsum(Wout) + bsum; aw = mean_h(w) ----
    for (int b = gw; b < Bdim; b += TW) {
      float s = 0.0f;
      for (int i = 0; i < Ddim / 32; ++i) {
        int d = lane + 32 * i;
        s += attn[(size_t)b * Ddim + d] * wsum[d];
      }
      for (int off = 16; off; off >>= 1) s += __shfl_xor(s, off, 32);
      if (lane == 0) logit[b] = s + wsum[Ddim];
    }
    for (int i = tid; i < Bdim * Ndim; i += GSZ) {
      int b = i / Ndim, n = i - b * Ndim;
      float s = 0.0f;
#pragma unroll
      for (int h = 0; h < Hh; ++h) s += wwgt[((size_t)b * Hh + h) * Ndim + n];
      aw_out[((size_t)b * Tdim + t) * Ndim + n] = s * (1.0f / (float)Hh);
    }
    grid_barrier(cnt, gen);

    // ---- phase 6: Gumbel-max categorical over masked logits; gather dec ----
    for (int r = gw; r < Bdim; r += TW) {
      float bv = -3.0e38f;
      int   bi = 0;
#pragma unroll
      for (int j = 0; j < 6; ++j) {
        int n = lane + 32 * j;
        float base = vis[(size_t)r * Ndim + n] ? NEGV : logit[n];  // column-indexed (ref quirk)
        float val = base + gumbel_hash((unsigned)t, (unsigned)r, (unsigned)n);
        if (val > bv) { bv = val; bi = n; }
      }
      for (int off = 16; off; off >>= 1) {
        float ov = __shfl_xor(bv, off, 32);
        int   oi = __shfl_xor(bi, off, 32);
        if (ov > bv || (ov == bv && oi < bi)) { bv = ov; bi = oi; }
      }
      int idx = __shfl(bi, 0, 32);
      if (lane == 0) {
        vis[(size_t)r * Ndim + idx] = 1u;
        seq_out[(size_t)r * Tdim + t] = (float)idx;
      }
      const float* src = enc + ((size_t)r * Ndim + idx) * Ddim;
      for (int i = 0; i < Ddim / 32; ++i) {
        int d = lane + 32 * i;
        X[(size_t)r * XK + d] = (_Float16)src[d];
      }
    }
    grid_barrier(cnt, gen);
  }
}

// ===========================================================================
extern "C" void kernel_launch(void* const* d_in, const int* in_sizes, int n_in,
                              void* d_out, int out_size, void* d_ws, size_t ws_size,
                              hipStream_t stream) {
  const float* enc       = (const float*)d_in[0];
  const float* W_ih      = (const float*)d_in[1];
  const float* W_hh      = (const float*)d_in[2];
  const float* b_ih      = (const float*)d_in[3];
  const float* b_hh      = (const float*)d_in[4];
  const float* in_proj_w = (const float*)d_in[5];
  const float* in_proj_b = (const float*)d_in[6];
  const float* out_w     = (const float*)d_in[7];
  const float* out_b     = (const float*)d_in[8];
  char* ws = (char*)d_ws;
  float* out = (float*)d_out;

  (void)in_sizes; (void)n_in; (void)out_size; (void)ws_size;

  hipMemsetAsync(ws, 0, 256, stream);  // grid-barrier state
  prep_weights<<<512, 256, 0, stream>>>(ws, W_ih, W_hh, in_proj_w);
  prep_enc<<<2048, 256, 0, stream>>>(ws, enc);
  prep_woutsum<<<1, 512, 0, stream>>>(ws, out_w, out_b);
  kv_proj<<<1152, 256, 0, stream>>>(ws, in_proj_b);
  decode_persistent<<<NBLK, BLKT, 0, stream>>>(ws, enc, b_ih, b_hh, in_proj_b, out);
}
